// MultiHeadedAttention_83038897701238
// MI455X (gfx1250) — compile-verified, matrix operand order checked by reading, not run on hardware
//
#include <hip/hip_runtime.h>
#include <hip/hip_bf16.h>

#define HID  1024
#define NH   16
#define HD   64
#define BB   4
#define SS   2048

typedef __bf16 bf16_t;
typedef __attribute__((ext_vector_type(16))) __bf16 v16bf;
typedef __attribute__((ext_vector_type(8)))  __bf16 v8bf;
typedef __attribute__((ext_vector_type(2)))  __bf16 v2bf;
typedef __attribute__((ext_vector_type(8)))  float  v8f;

static __device__ __forceinline__ v8f zero8() {
  v8f z = {0.f,0.f,0.f,0.f,0.f,0.f,0.f,0.f};
  return z;
}

// D = A(16x32 bf16) * B(32x16 bf16) + C(16x16 f32)
static __device__ __forceinline__ v8f wmma_bf16(v16bf a, v16bf b, v8f c) {
  return __builtin_amdgcn_wmma_f32_16x16x32_bf16(
      /*neg_a=*/false, a, /*neg_b=*/false, b,
      /*c_mod=*/(short)0, c, /*reuse_a=*/false, /*reuse_b=*/false);
}

static __device__ __forceinline__ v16bf cat16(v8bf lo, v8bf hi8) {
  return __builtin_shufflevector(lo, hi8,
      0,1,2,3,4,5,6,7,8,9,10,11,12,13,14,15);
}

static __device__ __forceinline__ v8bf ld8(const bf16_t* __restrict__ p) {
  return *reinterpret_cast<const v8bf*>(p);
}

// Raw 16-float (64B) staging for one 16-element bf16 fragment:
// two contiguous 8-float runs at k0+8*hi and k0+16+8*hi (ISA 7.12.2 layout).
struct RawF16 { float4 q0, q1, q2, q3; };

static __device__ __forceinline__ RawF16 ldraw(const float* __restrict__ row,
                                               int k0, int hi) {
  const float* p0 = row + k0 + 8*hi;
  const float* p1 = row + k0 + 16 + 8*hi;
  RawF16 r;
  r.q0 = *reinterpret_cast<const float4*>(p0);
  r.q1 = *reinterpret_cast<const float4*>(p0 + 4);
  r.q2 = *reinterpret_cast<const float4*>(p1);
  r.q3 = *reinterpret_cast<const float4*>(p1 + 4);
  return r;
}

static __device__ __forceinline__ v8bf cvt8v(float4 a, float4 b) {
  v8bf r;
  r[0]=(bf16_t)a.x; r[1]=(bf16_t)a.y; r[2]=(bf16_t)a.z; r[3]=(bf16_t)a.w;
  r[4]=(bf16_t)b.x; r[5]=(bf16_t)b.y; r[6]=(bf16_t)b.z; r[7]=(bf16_t)b.w;
  return r;  // lowers to v_cvt_pk_bf16_f32 x4
}

static __device__ __forceinline__ v16bf cvt16(const RawF16& r) {
  return cat16(cvt8v(r.q0, r.q1), cvt8v(r.q2, r.q3));
}

static __device__ __forceinline__ v16bf frag_bf(const bf16_t* __restrict__ row,
                                                int k0, int hi) {
  return cat16(ld8(row + k0 + 8*hi), ld8(row + k0 + 16 + 8*hi));
}

// Scheduling pipeline for one GEMM k-slab iteration:
// [NLOADS x VMEM-read] then (cvtA+cvtB0, wmma) (cvtB, wmma) x3.
// Masks: 0x20 = VMEM read, 0x2 = VALU, 0x8 = matrix (WMMA).
template<int NLOADS>
static __device__ __forceinline__ void sched_gemm_iter() {
  __builtin_amdgcn_sched_group_barrier(0x020, NLOADS, 0);
  __builtin_amdgcn_sched_group_barrier(0x002, 16, 0);
  __builtin_amdgcn_sched_group_barrier(0x008, 1, 0);
  __builtin_amdgcn_sched_group_barrier(0x002, 8, 0);
  __builtin_amdgcn_sched_group_barrier(0x008, 1, 0);
  __builtin_amdgcn_sched_group_barrier(0x002, 8, 0);
  __builtin_amdgcn_sched_group_barrier(0x008, 1, 0);
  __builtin_amdgcn_sched_group_barrier(0x002, 8, 0);
  __builtin_amdgcn_sched_group_barrier(0x008, 1, 0);
}

// ---------------------------------------------------------------------------
// Kernel 1: y = X @ W^T + b, optional fused RoPE, output bf16 [B, NH, S, HD].
// One wave = 16 tokens x one head (64 cols = 4 WMMA column tiles).
// RoPE column permutation: column c of tiles (0,1) holds the rotary pair
// (dim 2c, 2c+1) in the SAME lane; tiles (2,3) the pair (32+2c, 33+2c).
// Inner loop is double-buffered: slab i+1's 20 raw b128 loads are issued
// into a separate register set while slab i is converted and WMMA'd.
// ---------------------------------------------------------------------------
__global__ __launch_bounds__(256) void qkv_proj_rope_kernel(
    const float* __restrict__ X, const float* __restrict__ W,
    const float* __restrict__ bias, bf16_t* __restrict__ out, int do_rope)
{
  const int lane = threadIdx.x & 31;
  const int c    = lane & 15;
  const int hi   = lane >> 4;
  const int job  = blockIdx.x * 8 + (threadIdx.x >> 5);   // 8192 wave jobs
  const int head = job & (NH - 1);
  const int tok0 = (job >> 4) * 16;

  const float* xrow = X + (size_t)(tok0 + c) * HID;       // A row = lane&15

  int n0, n1, n2, n3;
  if (do_rope) {
    n0 = head*HD + 2*c;       n1 = n0 + 1;                // pair p = c
    n2 = head*HD + 32 + 2*c;  n3 = n2 + 1;                // pair p = 16+c
  } else {
    n0 = head*HD + c; n1 = n0 + 16; n2 = n0 + 32; n3 = n0 + 48;
  }
  const float* w0 = W + (size_t)n0 * HID;
  const float* w1 = W + (size_t)n1 * HID;
  const float* w2 = W + (size_t)n2 * HID;
  const float* w3 = W + (size_t)n3 * HID;

  v8f a0 = zero8(), a1 = zero8(), a2 = zero8(), a3 = zero8();

  // Prologue: load slab 0.
  RawF16 ra  = ldraw(xrow, 0, hi);
  RawF16 rb0 = ldraw(w0,   0, hi);
  RawF16 rb1 = ldraw(w1,   0, hi);
  RawF16 rb2 = ldraw(w2,   0, hi);
  RawF16 rb3 = ldraw(w3,   0, hi);

  for (int k0 = 32; k0 < HID; k0 += 32) {
    // Next slab's loads (independent registers -> stay in flight over WMMAs).
    RawF16 na  = ldraw(xrow, k0, hi);
    RawF16 nb0 = ldraw(w0,   k0, hi);
    RawF16 nb1 = ldraw(w1,   k0, hi);
    RawF16 nb2 = ldraw(w2,   k0, hi);
    RawF16 nb3 = ldraw(w3,   k0, hi);
    // Current slab: convert + WMMA.
    v16bf af = cvt16(ra);
    a0 = wmma_bf16(af, cvt16(rb0), a0);
    a1 = wmma_bf16(af, cvt16(rb1), a1);
    a2 = wmma_bf16(af, cvt16(rb2), a2);
    a3 = wmma_bf16(af, cvt16(rb3), a3);
    sched_gemm_iter<20>();
    ra = na; rb0 = nb0; rb1 = nb1; rb2 = nb2; rb3 = nb3;
  }
  {   // Tail slab.
    v16bf af = cvt16(ra);
    a0 = wmma_bf16(af, cvt16(rb0), a0);
    a1 = wmma_bf16(af, cvt16(rb1), a1);
    a2 = wmma_bf16(af, cvt16(rb2), a2);
    a3 = wmma_bf16(af, cvt16(rb3), a3);
  }

  const float b0 = bias[n0], b1 = bias[n1], b2 = bias[n2], b3 = bias[n3];
  const int  bidx = tok0 / SS;
  const int  s0r  = tok0 % SS;
  bf16_t* obase = out + (((size_t)bidx * NH + head) * SS + s0r) * HD;

  if (do_rope) {
    const float LOGB = 9.2103403719761836f;               // ln(10000)
    const float invf0 = __expf(-((float)c        / 32.f) * LOGB);
    const float invf1 = __expf(-((float)(16 + c) / 32.f) * LOGB);
#pragma unroll
    for (int r = 0; r < 8; ++r) {
      const int   m   = r + 8*hi;                         // C row = r + 8*hi
      const float pos = (float)(s0r + m);
      const float e0 = a0[r] + b0, o0 = a1[r] + b1;
      const float e1 = a2[r] + b2, o1 = a3[r] + b3;
      const float c0 = __cosf(pos * invf0), sn0 = __sinf(pos * invf0);
      const float c1 = __cosf(pos * invf1), sn1 = __sinf(pos * invf1);
      bf16_t* orow = obase + (size_t)m * HD;
      orow[c]      = (bf16_t)(e0*c0 - o0*sn0);
      orow[32 + c] = (bf16_t)(e0*sn0 + o0*c0);
      orow[16 + c] = (bf16_t)(e1*c1 - o1*sn1);
      orow[48 + c] = (bf16_t)(e1*sn1 + o1*c1);
    }
  } else {
#pragma unroll
    for (int r = 0; r < 8; ++r) {
      const int m = r + 8*hi;
      bf16_t* orow = obase + (size_t)m * HD;
      orow[c]      = (bf16_t)(a0[r] + b0);
      orow[16 + c] = (bf16_t)(a1[r] + b1);
      orow[32 + c] = (bf16_t)(a2[r] + b2);
      orow[48 + c] = (bf16_t)(a3[r] + b3);
    }
  }
}

// ---------------------------------------------------------------------------
// Kernel 2: flash attention. One wave = 16 queries x full head dim (4 f32
// accumulators). Streams 32-key tiles (causally bounded), online softmax,
// P transposed through per-wave LDS, V transposed through per-wave LDS with
// packed b32 stores (each lane owns a pair of adjacent keys).
// ---------------------------------------------------------------------------
__global__ __launch_bounds__(256) void attn_flash_kernel(
    const bf16_t* __restrict__ qws, const bf16_t* __restrict__ kws,
    const bf16_t* __restrict__ vws, const int* __restrict__ pad,
    const int* __restrict__ is_causal_p, bf16_t* __restrict__ hws)
{
  __shared__ __align__(16) bf16_t pbuf [8][16][40];   // P   16x32 (+pad)
  __shared__ __align__(16) bf16_t vtbuf[8][64][40];   // V^T 64x32 (+pad)

  const int lane = threadIdx.x & 31;
  const int wv   = threadIdx.x >> 5;
  const int c    = lane & 15;
  const int hi   = lane >> 4;

  const int bh   = blockIdx.x >> 4;        // 0..63
  const int qblk = blockIdx.x & 15;        // 0..15
  const int b    = bh >> 4;
  const int h    = bh & 15;
  const int qw   = qblk * 128 + wv * 16;   // this wave's 16 query rows

  const bf16_t* qbase = qws + ((size_t)bh * SS + qw) * HD;
  const bf16_t* kbase = kws + (size_t)bh * SS * HD;
  const bf16_t* vbase = vws + (size_t)bh * SS * HD;
  const int causal = *is_causal_p;
  const int* pm = pad + b * SS;

  // Preload Q fragments (A: row = query = lane&15, K-dim = head dim).
  const bf16_t* qrow = qbase + (size_t)c * HD;
  const v16bf qf0 = frag_bf(qrow, 0,  hi);
  const v16bf qf1 = frag_bf(qrow, 32, hi);

  v8f o0 = zero8(), o1 = zero8(), o2 = zero8(), o3 = zero8();
  float mrun[8], lrun[8];
#pragma unroll
  for (int r = 0; r < 8; ++r) { mrun[r] = -1e30f; lrun[r] = 0.f; }

  const int kend = causal ? (qw + 16) : SS;
  for (int jb = 0; jb < kend; jb += 32) {
    // --- batch all global loads for this tile up front --------------------
    const bf16_t* krow0 = kbase + (size_t)(jb + c)      * HD;
    const bf16_t* krow1 = kbase + (size_t)(jb + 16 + c) * HD;
    const v8bf k00a = ld8(krow0 +      8*hi), k00b = ld8(krow0 + 16 + 8*hi);
    const v8bf k01a = ld8(krow0 + 32 + 8*hi), k01b = ld8(krow0 + 48 + 8*hi);
    const v8bf k10a = ld8(krow1 +      8*hi), k10b = ld8(krow1 + 16 + 8*hi);
    const v8bf k11a = ld8(krow1 + 32 + 8*hi), k11b = ld8(krow1 + 48 + 8*hi);
    const int col0 = jb + c, col1 = jb + 16 + c;
    const int pm0 = pm[col0], pm1 = pm[col1];
    // V tile: lane owns keys (2c, 2c+1), dims [32*hi, 32*hi+32).
    const int kk = 2 * c;
    const bf16_t* vr0 = vbase + (size_t)(jb + kk)     * HD + 32*hi;
    const bf16_t* vr1 = vbase + (size_t)(jb + kk + 1) * HD + 32*hi;
    v8bf vA[4], vB[4];
#pragma unroll
    for (int dblk = 0; dblk < 4; ++dblk) {
      vA[dblk] = ld8(vr0 + 8*dblk);
      vB[dblk] = ld8(vr1 + 8*dblk);
    }

    // --- scores: S = Q (16x64) @ K^T (64x32) ------------------------------
    v8f s0 = zero8(), s1 = zero8();
    s0 = wmma_bf16(qf0, cat16(k00a, k00b), s0);
    s0 = wmma_bf16(qf1, cat16(k01a, k01b), s0);
    s1 = wmma_bf16(qf0, cat16(k10a, k10b), s1);
    s1 = wmma_bf16(qf1, cat16(k11a, k11b), s1);

    // --- V^T into per-wave LDS (packed 2-key b32 stores) ------------------
#pragma unroll
    for (int dblk = 0; dblk < 4; ++dblk) {
#pragma unroll
      for (int e = 0; e < 8; ++e) {
        const int d = 32*hi + dblk*8 + e;
        v2bf pkd; pkd[0] = vA[dblk][e]; pkd[1] = vB[dblk][e];
        *reinterpret_cast<v2bf*>(&vtbuf[wv][d][kk]) = pkd;
      }
    }

    // Pin: all 18 VMEM reads first, then the 4 score WMMAs; the V^T ds
    // stores and softmax VALU fill the WMMA shadow by dependency order.
    __builtin_amdgcn_sched_group_barrier(0x020, 18, 0);  // VMEM reads
    __builtin_amdgcn_sched_group_barrier(0x008, 4, 0);   // score WMMAs

    // --- online softmax per row (row = r + 8*hi) --------------------------
    float p0v[8], p1v[8];
#pragma unroll
    for (int r = 0; r < 8; ++r) {
      const int qrowi = qw + r + 8*hi;
      float v0 = s0[r] * 0.125f;                 // 1/sqrt(64)
      float v1 = s1[r] * 0.125f;
      if (pm0 == 0 || (causal && col0 > qrowi)) v0 = -1.0e9f;
      if (pm1 == 0 || (causal && col1 > qrowi)) v1 = -1.0e9f;
      float rm = fmaxf(v0, v1);
#pragma unroll
      for (int off = 1; off < 16; off <<= 1)     // stays within 16-lane half
        rm = fmaxf(rm, __shfl_xor(rm, off, 32));
      const float mnew  = fmaxf(mrun[r], rm);
      const float alpha = __expf(mrun[r] - mnew);
      const float e0 = __expf(v0 - mnew);
      const float e1 = __expf(v1 - mnew);
      float rs = e0 + e1;
#pragma unroll
      for (int off = 1; off < 16; off <<= 1)
        rs += __shfl_xor(rs, off, 32);
      lrun[r] = lrun[r] * alpha + rs;
      mrun[r] = mnew;
      o0[r] *= alpha; o1[r] *= alpha; o2[r] *= alpha; o3[r] *= alpha;
      p0v[r] = e0; p1v[r] = e1;
    }

    // --- P: C-layout -> A-layout via per-wave LDS -------------------------
#pragma unroll
    for (int r = 0; r < 8; ++r) {
      pbuf[wv][r + 8*hi][c]      = (bf16_t)p0v[r];
      pbuf[wv][r + 8*hi][16 + c] = (bf16_t)p1v[r];
    }
    asm volatile("s_wait_dscnt 0" ::: "memory");   // wave-private LDS RAW

    // --- O += P (16x32) @ V (32x64) ---------------------------------------
    const v16bf pA = frag_bf(&pbuf[wv][c][0], 0, hi);
    o0 = wmma_bf16(pA, frag_bf(&vtbuf[wv][ 0 + c][0], 0, hi), o0);
    o1 = wmma_bf16(pA, frag_bf(&vtbuf[wv][16 + c][0], 0, hi), o1);
    o2 = wmma_bf16(pA, frag_bf(&vtbuf[wv][32 + c][0], 0, hi), o2);
    o3 = wmma_bf16(pA, frag_bf(&vtbuf[wv][48 + c][0], 0, hi), o3);
  }

  // --- epilogue: O / l, write hidden bf16 [B, S, HID] ---------------------
#pragma unroll
  for (int r = 0; r < 8; ++r) {
    const int   m   = r + 8*hi;
    const float inv = 1.f / lrun[r];
    bf16_t* hrow = hws + ((size_t)b * SS + qw + m) * HID + h * HD;
    hrow[c]      = (bf16_t)(o0[r] * inv);
    hrow[16 + c] = (bf16_t)(o1[r] * inv);
    hrow[32 + c] = (bf16_t)(o2[r] * inv);
    hrow[48 + c] = (bf16_t)(o3[r] * inv);
  }
}

// ---------------------------------------------------------------------------
// Kernel 3: out = hidden(bf16) @ Wo^T + bo, f32 output. Double-buffered.
// ---------------------------------------------------------------------------
__global__ __launch_bounds__(256) void out_proj_kernel(
    const bf16_t* __restrict__ Hs, const float* __restrict__ W,
    const float* __restrict__ bias, float* __restrict__ out)
{
  const int lane = threadIdx.x & 31;
  const int c    = lane & 15;
  const int hi   = lane >> 4;
  const int job  = blockIdx.x * 8 + (threadIdx.x >> 5);
  const int ng   = job & 15;
  const int tok0 = (job >> 4) * 16;

  const bf16_t* hrow = Hs + (size_t)(tok0 + c) * HID;
  const int n0 = ng*64 + c, n1 = n0 + 16, n2 = n0 + 32, n3 = n0 + 48;
  const float* w0 = W + (size_t)n0 * HID;
  const float* w1 = W + (size_t)n1 * HID;
  const float* w2 = W + (size_t)n2 * HID;
  const float* w3 = W + (size_t)n3 * HID;

  v8f a0 = zero8(), a1 = zero8(), a2 = zero8(), a3 = zero8();

  v8bf ha  = ld8(hrow + 8*hi);
  v8bf hb  = ld8(hrow + 16 + 8*hi);
  RawF16 rb0 = ldraw(w0, 0, hi);
  RawF16 rb1 = ldraw(w1, 0, hi);
  RawF16 rb2 = ldraw(w2, 0, hi);
  RawF16 rb3 = ldraw(w3, 0, hi);

  for (int k0 = 32; k0 < HID; k0 += 32) {
    v8bf nha = ld8(hrow + k0 + 8*hi);
    v8bf nhb = ld8(hrow + k0 + 16 + 8*hi);
    RawF16 nb0 = ldraw(w0, k0, hi);
    RawF16 nb1 = ldraw(w1, k0, hi);
    RawF16 nb2 = ldraw(w2, k0, hi);
    RawF16 nb3 = ldraw(w3, k0, hi);
    const v16bf af = cat16(ha, hb);
    a0 = wmma_bf16(af, cvt16(rb0), a0);
    a1 = wmma_bf16(af, cvt16(rb1), a1);
    a2 = wmma_bf16(af, cvt16(rb2), a2);
    a3 = wmma_bf16(af, cvt16(rb3), a3);
    sched_gemm_iter<18>();
    ha = nha; hb = nhb; rb0 = nb0; rb1 = nb1; rb2 = nb2; rb3 = nb3;
  }
  {   // Tail slab.
    const v16bf af = cat16(ha, hb);
    a0 = wmma_bf16(af, cvt16(rb0), a0);
    a1 = wmma_bf16(af, cvt16(rb1), a1);
    a2 = wmma_bf16(af, cvt16(rb2), a2);
    a3 = wmma_bf16(af, cvt16(rb3), a3);
  }

  const float b0 = bias[n0], b1 = bias[n1], b2 = bias[n2], b3 = bias[n3];
#pragma unroll
  for (int r = 0; r < 8; ++r) {
    const int m = r + 8*hi;
    float* orow = out + (size_t)(tok0 + m) * HID;
    orow[n0] = a0[r] + b0;
    orow[n1] = a1[r] + b1;
    orow[n2] = a2[r] + b2;
    orow[n3] = a3[r] + b3;
  }
}

// ---------------------------------------------------------------------------
extern "C" void kernel_launch(void* const* d_in, const int* in_sizes, int n_in,
                              void* d_out, int out_size, void* d_ws, size_t ws_size,
                              hipStream_t stream) {
  const float* q   = (const float*)d_in[0];
  const float* k   = (const float*)d_in[1];
  const float* v   = (const float*)d_in[2];
  const int*   pad = (const int*)  d_in[3];
  const int*   isc = (const int*)  d_in[4];
  const float* Wq  = (const float*)d_in[5];
  const float* bq  = (const float*)d_in[6];
  const float* Wk  = (const float*)d_in[7];
  const float* bk  = (const float*)d_in[8];
  const float* Wv  = (const float*)d_in[9];
  const float* bv  = (const float*)d_in[10];
  const float* Wo  = (const float*)d_in[11];
  const float* bo  = (const float*)d_in[12];
  float* out = (float*)d_out;

  const size_t per = (size_t)BB * NH * SS * HD;   // elements per bf16 tensor
  bf16_t* qws = (bf16_t*)d_ws;
  bf16_t* kws = qws + per;
  bf16_t* vws = kws + per;
  bf16_t* hws = vws + per;                        // [B,S,HID] bf16

  const dim3 blk(256);
  const int gProj = (BB * SS / 16) * NH / 8;      // 1024 blocks
  const int gAttn = BB * NH * (SS / 128);         // 1024 blocks

  qkv_proj_rope_kernel<<<gProj, blk, 0, stream>>>(q, Wq, bq, qws, 1);
  qkv_proj_rope_kernel<<<gProj, blk, 0, stream>>>(k, Wk, bk, kws, 1);
  qkv_proj_rope_kernel<<<gProj, blk, 0, stream>>>(v, Wv, bv, vws, 0);
  attn_flash_kernel  <<<gAttn, blk, 0, stream>>>(qws, kws, vws, pad, isc, hws);
  out_proj_kernel    <<<gProj, blk, 0, stream>>>(hws, Wo, bo, out);
}
